// TTTBase_764504179384
// MI455X (gfx1250) — compile-verified, matrix operand order checked by reading, not run on hardware
//
#include <hip/hip_runtime.h>

// Problem constants (fixed by the reference)
#define B_    4
#define L_    2048
#define C_    2048
#define OUT_  768
#define NH_   32
#define HD_   64
#define MBS_  16
#define NM_   (L_ / MBS_)
#define EPS_  1e-6f
#define KSTEP 64

typedef __attribute__((ext_vector_type(16))) __bf16 v16bf;
typedef __attribute__((ext_vector_type(8)))  float  v8f;

__device__ __forceinline__ v8f wmma_bf16(v16bf a, v16bf b, v8f c) {
  // (neg_a, A, neg_b, B, c_mod, C, reuse_a, reuse_b) -> v_wmma_f32_16x16x32_bf16
  return __builtin_amdgcn_wmma_f32_16x16x32_bf16(false, a, false, b, (short)0, c,
                                                 false, false);
}

// bf16 16x16x32 operand fragment from K-contiguous (row-major) LDS data.
// Per ISA 7.12.2 the per-lane 16 elements are two contiguous 8-element runs:
// k in [hi*8, hi*8+8) and [16+hi*8, 16+hi*8+8)  ->  2x ds_load_b128.
__device__ __forceinline__ v16bf frag_a_bf(const __bf16* A, int lda, int lane) {
  int m = lane & 15, hi = lane >> 4;
  v16bf r;
#pragma unroll
  for (int v = 0; v < 8; ++v) {
    int k0 = ((v < 4) ? (2 * v) : (16 + 2 * (v - 4))) + hi * 8;
    r[2 * v]     = A[m * lda + k0];
    r[2 * v + 1] = A[m * lda + k0 + 1];
  }
  return r;
}

// --- CDNA5 async global->LDS copy (ASYNCcnt path), 16B per lane -------------
__device__ __forceinline__ void async_g2l_b128(unsigned int lds_off,
                                               const void* gptr) {
  asm volatile("global_load_async_to_lds_b128 %0, %1, off"
               :: "v"(lds_off), "v"((unsigned long long)gptr)
               : "memory");
}
__device__ __forceinline__ void wait_async0() {
  asm volatile("s_wait_asynccnt 0x0" ::: "memory");
}

// ---------------------------------------------------------------------------
// f32 -> bf16 cast (grid-stride)
// ---------------------------------------------------------------------------
__global__ void k_cvt_bf16(const float* __restrict__ src, __bf16* __restrict__ dst,
                           int n) {
  int i = blockIdx.x * blockDim.x + threadIdx.x;
  int stride = gridDim.x * blockDim.x;
  for (; i < n; i += stride) dst[i] = (__bf16)src[i];
}

// ---------------------------------------------------------------------------
// f32 (KxN) -> bf16 transposed (NxK): one pass, tiled through LDS
// ---------------------------------------------------------------------------
__global__ __launch_bounds__(256) void k_cvtT_bf16(const float* __restrict__ W,
                                                   __bf16* __restrict__ WT,
                                                   int K, int N) {
  __shared__ float tile[32][33];
  int nt = blockIdx.x * 32, kt = blockIdx.y * 32;
  int x = threadIdx.x, y = threadIdx.y;
#pragma unroll
  for (int i = 0; i < 4; ++i)
    tile[y + 8 * i][x] = W[(size_t)(kt + y + 8 * i) * N + nt + x];
  __syncthreads();
#pragma unroll
  for (int i = 0; i < 4; ++i)
    WT[(size_t)(nt + y + 8 * i) * K + kt + x] = (__bf16)tile[x][y + 8 * i];
}

// ---------------------------------------------------------------------------
// WMMA GEMM: D(MxN) = A(MxK,bf16 row-major) @ B, with B given TRANSPOSED
// (BT is NxK bf16 row-major) so both operands fragment as K-contiguous b128.
// 8 waves, workgroup tile 128x64, wave tile 32x32, K-step 64, double-buffered
// async global->LDS staging.
// mode 0: row-major f32 out; mode 1: scan layout (B,NH,L,HD) + RoPE; mode 2: no RoPE
// ---------------------------------------------------------------------------
__global__ __launch_bounds__(256) void k_gemm_bf16(
    const __bf16* __restrict__ A, const __bf16* __restrict__ BT,
    float* __restrict__ D, int M, int N, int K, int mode) {
  __shared__ __bf16 sA[2][128 * KSTEP];
  __shared__ __bf16 sB[2][64 * KSTEP];

  int t = threadIdx.x, lane = t & 31, w = t >> 5;
  int wm = w >> 1, wn = w & 1;                 // 4x2 wave grid
  int m0b = blockIdx.y * 128, n0b = blockIdx.x * 64;

  v8f acc[2][2];
#pragma unroll
  for (int i = 0; i < 2; ++i)
#pragma unroll
    for (int j = 0; j < 2; ++j)
#pragma unroll
      for (int e = 0; e < 8; ++e) acc[i][j][e] = 0.f;

  int ar = t >> 1, ac = (t & 1) * 32;          // A: 128 rows x 64 k, 32 bf16/thread
  int br = t >> 2, bc = (t & 3) * 16;          // BT: 64 rows x 64 k, 16 bf16/thread
  const __bf16* gA = A + (size_t)(m0b + ar) * K + ac;
  const __bf16* gB = BT + (size_t)(n0b + br) * K + bc;

  auto stage = [&](int buf, int kt) {
#pragma unroll
    for (int c = 0; c < 4; ++c)
      async_g2l_b128((unsigned int)(unsigned long long)&sA[buf][ar * KSTEP + ac + 8 * c],
                     gA + kt + 8 * c);
#pragma unroll
    for (int c = 0; c < 2; ++c)
      async_g2l_b128((unsigned int)(unsigned long long)&sB[buf][br * KSTEP + bc + 8 * c],
                     gB + kt + 8 * c);
  };

  stage(0, 0);
  wait_async0();
  __syncthreads();

  int buf = 0;
  for (int kt = 0; kt < K; kt += KSTEP) {
    if (kt + KSTEP < K) stage(buf ^ 1, kt + KSTEP);  // overlap with compute
#pragma unroll
    for (int kc = 0; kc < KSTEP / 32; ++kc) {
      v16bf af[2], bfr[2];
#pragma unroll
      for (int i = 0; i < 2; ++i)
        af[i] = frag_a_bf(&sA[buf][(wm * 32 + i * 16) * KSTEP + kc * 32], KSTEP, lane);
#pragma unroll
      for (int j = 0; j < 2; ++j)
        bfr[j] = frag_a_bf(&sB[buf][(wn * 32 + j * 16) * KSTEP + kc * 32], KSTEP, lane);
#pragma unroll
      for (int i = 0; i < 2; ++i)
#pragma unroll
        for (int j = 0; j < 2; ++j)
          acc[i][j] = wmma_bf16(af[i], bfr[j], acc[i][j]);
    }
    wait_async0();
    __syncthreads();
    buf ^= 1;
  }

  int hi = lane >> 4, n15 = lane & 15;
#pragma unroll
  for (int i = 0; i < 2; ++i) {
#pragma unroll
    for (int j = 0; j < 2; ++j) {
#pragma unroll
      for (int v = 0; v < 8; ++v) {
        int mrow = m0b + wm * 32 + i * 16 + v + 8 * hi;
        int ncol = n0b + wn * 32 + j * 16 + n15;
        float val = acc[i][j][v];
        if (mode == 1) {
          // interleaved RoPE: pair (2k,2k+1) within head dim, pos = token%16
          int d = ncol & 63;
          int pr = d >> 1;
          int pos = mrow & 15;
          float inv = __expf(-(float)(2 * pr) * (1.f / 64.f) * 9.210340371976184f);
          float th = (float)pos * inv;
          float cs = __cosf(th), sn = __sinf(th);
          float p = __shfl_xor(val, 1, 32);
          val = (d & 1) ? (val * cs + p * sn) : (val * cs - p * sn);
        }
        if (mode >= 1) {
          int bb = mrow >> 11;
          int l  = mrow & (L_ - 1);
          int hh = ncol >> 6;
          int dd = ncol & 63;
          D[(((size_t)bb * NH_ + hh) * L_ + l) * HD_ + dd] = val;
        } else {
          D[(size_t)mrow * N + ncol] = val;
        }
      }
    }
  }
}

// ---------------------------------------------------------------------------
// ttt_lr: lrs[(b*NH+h)*L + l] = sigmoid(x . w_h + bias_h) / HD ; wave/token
// ---------------------------------------------------------------------------
__global__ __launch_bounds__(32) void k_lr(const float* __restrict__ hs,
                                           const float* __restrict__ lrw,
                                           const float* __restrict__ lrb,
                                           float* __restrict__ lrs) {
  int tok = blockIdx.x;
  int h = threadIdx.x;
  const float4* x = (const float4*)(hs + (size_t)tok * C_);
  const float4* wv = (const float4*)(lrw + (size_t)h * C_);
  float s = 0.f;
  for (int c = 0; c < C_ / 4; ++c) {
    float4 a = x[c], b = wv[c];
    s += a.x * b.x + a.y * b.y + a.z * b.z + a.w * b.w;
  }
  s += lrb[h];
  float sig = 1.f / (1.f + __expf(-s));
  int b = tok / L_, l = tok - b * L_;
  lrs[((size_t)b * NH_ + h) * L_ + l] = sig * (1.f / (float)HD_);
}

// ---------------------------------------------------------------------------
// TTT scan: one workgroup (4 waves) per (b,h) chain. f32 W1/b1 master state in
// LDS; bf16 staged copies in fragment-friendly (K-contiguous) layouts so every
// WMMA operand loads as 2x ds_load_b128. K=16 matmuls are padded to 32 with
// zero regions written once.
// ---------------------------------------------------------------------------
__global__ __launch_bounds__(128) void k_scan(
    const float* __restrict__ XQ, const float* __restrict__ XK,
    const float* __restrict__ XV, const float* __restrict__ lrs,
    const float* __restrict__ W1g, const float* __restrict__ b1g,
    const float* __restrict__ ltok, const float* __restrict__ lnw,
    const float* __restrict__ lnb, float* __restrict__ so) {
  __shared__ float  sW1[64 * 64];     // master state, row-major
  __shared__ __bf16 sW1T[64 * 64];    // sW1T[n*64+k] = W1[k][n]  (B operand)
  __shared__ float  sXq[1024], sXk[1024], sXv[1024];
  __shared__ __bf16 sXqb[1024], sXkb[1024];   // row-major m*64+k
  __shared__ float  sZ[1024], sG[1024];
  __shared__ __bf16 sGT[64 * 32];     // sGT[n*32+k] = grad[k][n], k>=16 zero
  __shared__ __bf16 sMcb[16 * 32];    // -Mcomb row-major, k>=16 zero
  __shared__ __bf16 sUa[64 * 32];     // sUa[d*32+k] = -le*lr[k]*xk[k][d], k>=16 zero
  __shared__ float  sB1[64], sGam[64], sBet[64], sLr[16], sTok[16];

  int bh = blockIdx.x, b = bh >> 5, h = bh & 31;
  int t = threadIdx.x, lane = t & 31, w = t >> 5;
  int n15 = lane & 15, hi = lane >> 4;
  int r = t >> 3, c0 = (t & 7) * 8;   // row-wise LN partition: 8 threads/row

  for (int i = t; i < 4096; i += 128) {
    float v = W1g[h * 4096 + i];
    sW1[i] = v;
    sW1T[(i & 63) * 64 + (i >> 6)] = (__bf16)v;
  }
  for (int i = t; i < 64 * 16; i += 128) sGT[(i >> 4) * 32 + 16 + (i & 15)] = (__bf16)0.f;
  for (int i = t; i < 64 * 16; i += 128) sUa[(i >> 4) * 32 + 16 + (i & 15)] = (__bf16)0.f;
  for (int i = t; i < 16 * 16; i += 128) sMcb[(i >> 4) * 32 + 16 + (i & 15)] = (__bf16)0.f;
  if (t < 64) {
    sB1[t] = b1g[h * 64 + t];
    sGam[t] = lnw[h * 64 + t];
    sBet[t] = lnb[h * 64 + t];
  }
  if (t < 16) {
    float v = 1.f / (float)(t + 1) + ltok[t];
    sTok[t] = fmaxf(v, 0.f);
  }
  __syncthreads();

  const size_t base = (size_t)bh * (L_ * HD_);

  for (int n = 0; n < NM_; ++n) {
    // ---- load minibatch tiles (+ bf16 copies, + next-step prefetch) -----
    for (int i = t; i < 1024; i += 128) {
      size_t g = base + (size_t)n * 1024 + i;
      float q = XQ[g], k = XK[g], vv = XV[g];
      sXq[i] = q; sXk[i] = k; sXv[i] = vv;
      sXqb[i] = (__bf16)q; sXkb[i] = (__bf16)k;
    }
    if (t < 16) sLr[t] = lrs[(size_t)bh * L_ + n * 16 + t];
    if (n + 1 < NM_) {
      size_t gn = base + (size_t)(n + 1) * 1024 + t * 8;
      __builtin_prefetch(&XQ[gn], 0, 1);
      __builtin_prefetch(&XK[gn], 0, 1);
      __builtin_prefetch(&XV[gn], 0, 1);
    }
    __syncthreads();

    // ---- Z1 = xk @ W1 + b1 : wave w owns cols [16w,16w+16) --------------
    {
      v8f c;
      float bv = sB1[w * 16 + n15];
#pragma unroll
      for (int e = 0; e < 8; ++e) c[e] = bv;
#pragma unroll
      for (int kc = 0; kc < 2; ++kc) {
        v16bf a  = frag_a_bf(&sXkb[kc * 32], 64, lane);
        v16bf bb = frag_a_bf(&sW1T[(w * 16) * 64 + kc * 32], 64, lane);
        c = wmma_bf16(a, bb, c);
      }
#pragma unroll
      for (int v = 0; v < 8; ++v) sZ[(v + 8 * hi) * 64 + w * 16 + n15] = c[v];
    }
    __syncthreads();

    // ---- grad = ln_fused_l2_bwd(Z1, xv - xk); store f32 + padded bf16^T -
    {
      float z[8], xh[8], g[8];
      float s1 = 0.f;
#pragma unroll
      for (int q = 0; q < 8; ++q) { z[q] = sZ[r * 64 + c0 + q]; s1 += z[q]; }
#pragma unroll
      for (int o = 1; o < 8; o <<= 1) s1 += __shfl_xor(s1, o, 32);
      float mu = s1 * (1.f / 64.f);
      float s2 = 0.f;
#pragma unroll
      for (int q = 0; q < 8; ++q) { float d = z[q] - mu; s2 += d * d; }
#pragma unroll
      for (int o = 1; o < 8; o <<= 1) s2 += __shfl_xor(s2, o, 32);
      float rstd = rsqrtf(s2 * (1.f / 64.f) + EPS_);
      float sg = 0.f, sgx = 0.f;
#pragma unroll
      for (int q = 0; q < 8; ++q) {
        xh[q] = (z[q] - mu) * rstd;
        float gam = sGam[c0 + q];
        float y = gam * xh[q] + sBet[c0 + q];
        float tgt = sXv[r * 64 + c0 + q] - sXk[r * 64 + c0 + q];
        g[q] = (y - tgt) * gam;
        sg += g[q]; sgx += g[q] * xh[q];
      }
#pragma unroll
      for (int o = 1; o < 8; o <<= 1) {
        sg += __shfl_xor(sg, o, 32);
        sgx += __shfl_xor(sgx, o, 32);
      }
#pragma unroll
      for (int q = 0; q < 8; ++q) {
        float gr = (64.f * g[q] - sg - xh[q] * sgx) * rstd * (1.f / 64.f);
        sG[r * 64 + c0 + q] = gr;
        sGT[(c0 + q) * 32 + r] = (__bf16)gr;   // transposed B operand, k=r<16
      }
    }
    __syncthreads();

    // ---- T = xq @ W1 (all waves); wave0: Attn -> -Mcomb (padded bf16) ---
    v8f tAcc;
    {
#pragma unroll
      for (int e = 0; e < 8; ++e) tAcc[e] = 0.f;
#pragma unroll
      for (int kc = 0; kc < 2; ++kc) {
        v16bf a  = frag_a_bf(&sXqb[kc * 32], 64, lane);
        v16bf bb = frag_a_bf(&sW1T[(w * 16) * 64 + kc * 32], 64, lane);
        tAcc = wmma_bf16(a, bb, tAcc);
      }
    }
    if (w == 0) {
      v8f at;
#pragma unroll
      for (int e = 0; e < 8; ++e) at[e] = 0.f;
#pragma unroll
      for (int kc = 0; kc < 2; ++kc) {
        v16bf a  = frag_a_bf(&sXqb[kc * 32], 64, lane);
        v16bf bb = frag_a_bf(&sXkb[kc * 32], 64, lane);  // B[k][j] = xk[j][k]
        at = wmma_bf16(a, bb, at);
      }
#pragma unroll
      for (int v = 0; v < 8; ++v) {
        int mi = v + 8 * hi, nj = n15;
        float mc = (nj <= mi) ? sTok[mi] * sLr[nj] * (1.f + at[v]) : 0.f;
        sMcb[mi * 32 + nj] = (__bf16)(-mc);
      }
    }
    __syncthreads();

    // ---- Z1_bar = T + b1 - Mcomb @ grad (K=16 padded) -> reuse sZ -------
    {
      v8f c = tAcc;
      float bv = sB1[w * 16 + n15];
#pragma unroll
      for (int e = 0; e < 8; ++e) c[e] += bv;
      v16bf a  = frag_a_bf(sMcb, 32, lane);
      v16bf bb = frag_a_bf(&sGT[(w * 16) * 32], 32, lane);
      c = wmma_bf16(a, bb, c);
#pragma unroll
      for (int v = 0; v < 8; ++v) sZ[(v + 8 * hi) * 64 + w * 16 + n15] = c[v];
    }
    __syncthreads();

    // ---- out = xq + ln_fwd(Z1_bar); also build W1-update A tile ---------
    {
      float z[8];
      float s1 = 0.f;
#pragma unroll
      for (int q = 0; q < 8; ++q) { z[q] = sZ[r * 64 + c0 + q]; s1 += z[q]; }
#pragma unroll
      for (int o = 1; o < 8; o <<= 1) s1 += __shfl_xor(s1, o, 32);
      float mu = s1 * (1.f / 64.f);
      float s2 = 0.f;
#pragma unroll
      for (int q = 0; q < 8; ++q) { float d = z[q] - mu; s2 += d * d; }
#pragma unroll
      for (int o = 1; o < 8; o <<= 1) s2 += __shfl_xor(s2, o, 32);
      float rstd = rsqrtf(s2 * (1.f / 64.f) + EPS_);
      size_t orow = ((size_t)b * L_ + n * 16 + r) * C_ + h * 64 + c0;
#pragma unroll
      for (int q = 0; q < 8; ++q)
        so[orow + q] = sXq[r * 64 + c0 + q] +
                       sGam[c0 + q] * (z[q] - mu) * rstd + sBet[c0 + q];

      float le = sTok[15];
      int d = t >> 1;
#pragma unroll
      for (int q = 0; q < 8; ++q) {
        int k = (t & 1) * 8 + q;
        sUa[d * 32 + k] = (__bf16)(-le * sLr[k] * sXk[k * 64 + d]);
      }
    }
    __syncthreads();

    // ---- W1 -= (le*xk)^T @ grad ; b1 -= sum(le*grad) --------------------
    {
      v16bf ua = frag_a_bf(&sUa[(w * 16) * 32], 32, lane);
#pragma unroll
      for (int tt = 0; tt < 4; ++tt) {  // wave w owns W1 rows [16w,16w+16)
        v8f c;
#pragma unroll
        for (int v = 0; v < 8; ++v)
          c[v] = sW1[(w * 16 + v + 8 * hi) * 64 + tt * 16 + n15];
        v16bf bb = frag_a_bf(&sGT[(tt * 16) * 32], 32, lane);
        c = wmma_bf16(ua, bb, c);
#pragma unroll
        for (int v = 0; v < 8; ++v) {
          int row = w * 16 + v + 8 * hi, col = tt * 16 + n15;
          sW1[row * 64 + col] = c[v];
          sW1T[col * 64 + row] = (__bf16)c[v];
        }
      }
      if (t < 64) {
        float le = sTok[15];
        float s = 0.f;
#pragma unroll
        for (int j = 0; j < 16; ++j) s += le * sLr[j] * sG[j * 64 + t];
        sB1[t] -= s;
      }
    }
    __syncthreads();
  }
}

// ---------------------------------------------------------------------------
// post-norm over C=2048 -> bf16
// ---------------------------------------------------------------------------
__global__ __launch_bounds__(256) void k_postnorm(const float* __restrict__ x,
                                                  const float* __restrict__ pw,
                                                  const float* __restrict__ pb,
                                                  __bf16* __restrict__ y) {
  __shared__ float red[8];
  int row = blockIdx.x, t = threadIdx.x, lane = t & 31, w = t >> 5;
  const float* xr = x + (size_t)row * C_;
  float v[8];
  float s = 0.f;
#pragma unroll
  for (int q = 0; q < 8; ++q) { v[q] = xr[t * 8 + q]; s += v[q]; }
#pragma unroll
  for (int o = 1; o < 32; o <<= 1) s += __shfl_xor(s, o, 32);
  if (lane == 0) red[w] = s;
  __syncthreads();
  s = 0.f;
  for (int i = 0; i < 8; ++i) s += red[i];
  float mu = s / (float)C_;
  float s2 = 0.f;
#pragma unroll
  for (int q = 0; q < 8; ++q) { float d = v[q] - mu; s2 += d * d; }
#pragma unroll
  for (int o = 1; o < 32; o <<= 1) s2 += __shfl_xor(s2, o, 32);
  __syncthreads();
  if (lane == 0) red[w] = s2;
  __syncthreads();
  s2 = 0.f;
  for (int i = 0; i < 8; ++i) s2 += red[i];
  float rstd = rsqrtf(s2 / (float)C_ + EPS_);
#pragma unroll
  for (int q = 0; q < 8; ++q) {
    int c = t * 8 + q;
    y[(size_t)row * C_ + c] = (__bf16)((v[q] - mu) * rstd * pw[c] + pb[c]);
  }
}

// ---------------------------------------------------------------------------
extern "C" void kernel_launch(void* const* d_in, const int* in_sizes, int n_in,
                              void* d_out, int out_size, void* d_ws,
                              size_t ws_size, hipStream_t stream) {
  (void)in_sizes; (void)n_in; (void)out_size; (void)ws_size;
  const float* hs   = (const float*)d_in[0];
  // d_in[1] position_ids: arange broadcast -> pos%16 == token%16, derived on GPU
  const float* Wq   = (const float*)d_in[2];
  const float* Wk   = (const float*)d_in[3];
  const float* Wv   = (const float*)d_in[4];
  const float* Wo   = (const float*)d_in[5];
  const float* W1   = (const float*)d_in[6];
  const float* b1   = (const float*)d_in[7];
  const float* lrw  = (const float*)d_in[8];
  const float* lrb  = (const float*)d_in[9];
  const float* ltok = (const float*)d_in[10];
  const float* lnw  = (const float*)d_in[11];
  const float* lnb  = (const float*)d_in[12];
  const float* pw   = (const float*)d_in[13];
  const float* pb   = (const float*)d_in[14];

  char* ws = (char*)d_ws;
  size_t off = 0;
  auto alloc = [&](size_t bytes) {
    void* p = ws + off;
    off += (bytes + 255) & ~(size_t)255;
    return p;
  };
  __bf16* hsb = (__bf16*)alloc((size_t)B_ * L_ * C_ * 2);
  __bf16* wqT = (__bf16*)alloc((size_t)C_ * C_ * 2);
  __bf16* wkT = (__bf16*)alloc((size_t)C_ * C_ * 2);
  __bf16* wvT = (__bf16*)alloc((size_t)C_ * C_ * 2);
  __bf16* woT = (__bf16*)alloc((size_t)C_ * OUT_ * 2);
  float* xq  = (float*)alloc((size_t)B_ * NH_ * L_ * HD_ * 4);
  float* xk  = (float*)alloc((size_t)B_ * NH_ * L_ * HD_ * 4);
  float* xv  = (float*)alloc((size_t)B_ * NH_ * L_ * HD_ * 4);
  float* lrs = (float*)alloc((size_t)B_ * NH_ * L_ * 4);
  float* so  = (float*)alloc((size_t)B_ * L_ * C_ * 4);
  __bf16* nrm = (__bf16*)alloc((size_t)B_ * L_ * C_ * 2);

  // Cast hidden to bf16; cast+transpose weights (one pass each)
  k_cvt_bf16<<<1024, 256, 0, stream>>>(hs, hsb, B_ * L_ * C_);
  dim3 tT(32, 8);
  k_cvtT_bf16<<<dim3(C_ / 32, C_ / 32), tT, 0, stream>>>(Wq, wqT, C_, C_);
  k_cvtT_bf16<<<dim3(C_ / 32, C_ / 32), tT, 0, stream>>>(Wk, wkT, C_, C_);
  k_cvtT_bf16<<<dim3(C_ / 32, C_ / 32), tT, 0, stream>>>(Wv, wvT, C_, C_);
  k_cvtT_bf16<<<dim3(OUT_ / 32, C_ / 32), tT, 0, stream>>>(Wo, woT, C_, OUT_);

  // QKV projections with fused RoPE epilogue into scan layout (B,NH,L,HD)
  dim3 gQKV(C_ / 64, (B_ * L_) / 128);
  k_gemm_bf16<<<gQKV, 256, 0, stream>>>(hsb, wqT, xq, B_ * L_, C_, C_, 1);
  k_gemm_bf16<<<gQKV, 256, 0, stream>>>(hsb, wkT, xk, B_ * L_, C_, C_, 1);
  k_gemm_bf16<<<gQKV, 256, 0, stream>>>(hsb, wvT, xv, B_ * L_, C_, C_, 2);

  // Per-head learning rates
  k_lr<<<B_ * L_, 32, 0, stream>>>(hs, lrw, lrb, lrs);

  // Sequential TTT scan: one workgroup per (b,h)
  k_scan<<<B_ * NH_, 128, 0, stream>>>(xq, xk, xv, lrs, W1, b1, ltok, lnw, lnb, so);

  // Post-norm + output projection
  k_postnorm<<<B_ * L_, 256, 0, stream>>>(so, pw, pb, nrm);
  dim3 gO(OUT_ / 64, (B_ * L_) / 128);
  k_gemm_bf16<<<gO, 256, 0, stream>>>(nrm, woT, (float*)d_out, B_ * L_, OUT_, C_, 0);
}
